// HGConv_12644383719477
// MI455X (gfx1250) — matrix operand order repaired and checked by reading, not compiled, not run on hardware
//
#include <hip/hip_runtime.h>

typedef __attribute__((ext_vector_type(16))) _Float16 v16h;
typedef __attribute__((ext_vector_type(8)))  float    v8f;

#define NN      50000   // nodes
#define NE      50000   // hyperedges
#define NNZ_    500000
#define NG      50
#define INC     128
#define HIDC    256
#define OUTC    2

__device__ __forceinline__ void atomAddF(float* p, float v) {
  unsafeAtomicAdd(p, v);   // lowers to global_atomic_add_f32 on gfx1250
}

// ---------------- utility kernels ----------------
__global__ void k_zero(float* __restrict__ p, int n4) {
  int i = blockIdx.x * blockDim.x + threadIdx.x;
  if (i < n4) ((float4*)p)[i] = make_float4(0.f, 0.f, 0.f, 0.f);
}

__global__ void k_count(const int* __restrict__ idx, float* __restrict__ deg, int n) {
  int i = blockIdx.x * blockDim.x + threadIdx.x;
  if (i < n) atomAddF(&deg[idx[i]], 1.0f);
}

__global__ void k_inv(float* __restrict__ deg, int n) {
  int i = blockIdx.x * blockDim.x + threadIdx.x;
  if (i < n) { float d = deg[i]; deg[i] = d > 0.f ? 1.0f / d : 0.f; }
}

// ---------------- W -> WMMA B-fragment packed f16 ----------------
// Layout: for k-block kb (32 rows), n-tile nt (16 cols), lane (32), half h (16):
//   Wp[((kb*(N/16)+nt)*32 + lane)*16 + h] = W[kb*32 + (lane>>4)*16 + h][nt*16 + (lane&15)]
// so each lane's 16-half B fragment is one contiguous 32-byte chunk.
__global__ void k_pack_w(const float* __restrict__ W, _Float16* __restrict__ Wp,
                         int K, int N) {
  int idx = blockIdx.x * blockDim.x + threadIdx.x;
  int ntiles = N >> 4;
  int total = (K >> 5) * ntiles * 512;
  if (idx >= total) return;
  int h    = idx & 15;
  int lane = (idx >> 4) & 31;
  int t    = idx >> 9;
  int nt   = t % ntiles;
  int kb   = t / ntiles;
  int k = (kb << 5) + ((lane >> 4) << 4) + h;
  int n = (nt << 4) + (lane & 15);
  Wp[idx] = (_Float16)W[k * N + n];
}

// ---------------- WMMA GEMM: Y[M,N] = X[M,K] (f32) * W (packed f16) ----------------
// Block = 128 threads (4 waves). Block covers 64 M rows x 128 N cols.
// Each wave: 16(M) x 128(N) tile = 8 f32 accumulators, A fragment reused 8x.
// B strip for the whole block is staged once in LDS (one barrier), then
// WMMAs are fed from ds_load_b128 instead of global.
__global__ __launch_bounds__(128) void k_gemm(const float* __restrict__ X,
                                              const _Float16* __restrict__ Wp,
                                              float* __restrict__ Y,
                                              int M, int K, int N) {
  // LDS B stage: (K/32 <= 8) k-blocks * 8 n-tiles * 512 halves = 32768 halves (64KB)
  __shared__ _Float16 Bs[32768];

  int tid   = threadIdx.x;
  int wave  = tid >> 5;
  int lane  = tid & 31;
  int mtile = blockIdx.x * 4 + wave;
  int n0    = blockIdx.y * 128;
  int nt0   = n0 >> 4;            // first of 8 n-tiles for this block
  int ntiles = N >> 4;
  int kblocks = K >> 5;

  // cooperative global->LDS copy of the B strip (32B chunks)
  int nchunks = kblocks * 8 * 32;                  // chunk = 16 halves = 32B
  for (int c = tid; c < nchunks; c += 128) {
    int kbj  = c >> 5;                              // kb*8 + j
    int kb   = kbj >> 3;
    int j    = kbj & 7;
    int ln   = c & 31;
    const float4* src = (const float4*)(Wp + ((size_t)((kb * ntiles) + nt0 + j) * 32 + ln) * 16);
    float4* dst = (float4*)(Bs + (size_t)c * 16);
    dst[0] = src[0];
    dst[1] = src[1];
  }
  __syncthreads();

  if (mtile * 16 >= M) return;
  int m0 = mtile * 16;
  int r  = lane & 15;
  int hi = lane >> 4;

  const float* xrow = X + (size_t)(m0 + r) * K + hi * 16;

  v8f acc[8];
  #pragma unroll
  for (int j = 0; j < 8; ++j) acc[j] = (v8f){};

  for (int kb = 0; kb < kblocks; ++kb) {
    // A fragment (contiguous k order, matching packed B): one contiguous 16-float row segment/lane
    const float* xp = xrow + kb * 32;
    v16h a;
    #pragma unroll
    for (int i = 0; i < 8; ++i) a[i]     = (_Float16)xp[i];
    #pragma unroll
    for (int i = 0; i < 8; ++i) a[8 + i] = (_Float16)xp[8 + i];

    const _Float16* bb = Bs + ((size_t)(kb * 8) * 32 + lane) * 16;
    #pragma unroll
    for (int j = 0; j < 8; ++j) {
      v16h b = *(const v16h*)(bb + (size_t)j * 512);
      acc[j] = __builtin_amdgcn_wmma_f32_16x16x32_f16(
          false, a, false, b, (short)0, acc[j], false, false);
    }
  }

  // C/D layout: VGPR v, lane L: row = m0 + v + 8*(L>>4), col = n0 + j*16 + (L&15)
  float* yb = Y + (size_t)(m0 + hi * 8) * N + n0 + r;
  #pragma unroll
  for (int v = 0; v < 8; ++v) {
    #pragma unroll
    for (int j = 0; j < 8; ++j) yb[(size_t)v * N + j * 16] = acc[j][v];
  }
}

// ---------------- sparse propagation: dst[sidx[i]] += src[gidx[i]] ----------------
__global__ __launch_bounds__(256) void k_scatter(const float* __restrict__ src,
                                                 float* __restrict__ dst,
                                                 const int* __restrict__ gidx,
                                                 const int* __restrict__ sidx,
                                                 int nnz) {
  int w = blockIdx.x * 8 + (threadIdx.x >> 5);
  if (w >= nnz) return;
  int lane = threadIdx.x & 31;
  int g = gidx[w], s = sidx[w];
  const float* sp = src + (size_t)g * HIDC;
  float*       dp = dst + (size_t)s * HIDC;
  #pragma unroll
  for (int it = 0; it < 2; ++it) {
    int c = it * 128 + lane * 4;
    float4 v = *(const float4*)(sp + c);
    atomAddF(dp + c + 0, v.x);
    atomAddF(dp + c + 1, v.y);
    atomAddF(dp + c + 2, v.z);
    atomAddF(dp + c + 3, v.w);
  }
}

__global__ void k_scale_rows(float* __restrict__ e, const float* __restrict__ binv,
                             int rows) {
  int i = blockIdx.x * blockDim.x + threadIdx.x;
  if (i < rows * HIDC) e[i] *= binv[i >> 8];
}

__global__ void k_finish(float* __restrict__ h, const float* __restrict__ dinv,
                         const float* __restrict__ bias, int rows) {
  int i = blockIdx.x * blockDim.x + threadIdx.x;
  if (i >= rows * HIDC) return;
  float v = h[i] * dinv[i >> 8] + bias[i & (HIDC - 1)];
  h[i] = v > 0.f ? v : 0.f;
}

// ---------------- pooling ----------------
__global__ __launch_bounds__(256) void k_pool(const float* __restrict__ h,
                                              const int* __restrict__ batch,
                                              float* __restrict__ pooled) {
  int n = blockIdx.x * 8 + (threadIdx.x >> 5);
  if (n >= NN) return;
  int lane = threadIdx.x & 31;
  int b = batch[n];
  const float* sp = h + (size_t)n * HIDC;
  float*       dp = pooled + (size_t)b * HIDC;
  #pragma unroll
  for (int it = 0; it < 2; ++it) {
    int c = it * 128 + lane * 4;
    float4 v = *(const float4*)(sp + c);
    atomAddF(dp + c + 0, v.x);
    atomAddF(dp + c + 1, v.y);
    atomAddF(dp + c + 2, v.z);
    atomAddF(dp + c + 3, v.w);
  }
}

__global__ void k_pool_div(float* __restrict__ pooled, const float* __restrict__ cnt) {
  int i = blockIdx.x * blockDim.x + threadIdx.x;
  if (i >= NG * HIDC) return;
  float c = cnt[i >> 8];
  pooled[i] /= (c > 1.f ? c : 1.f);
}

// ---------------- MLP head (tiny) ----------------
__global__ void k_mlp1(const float* __restrict__ pooled, const float* __restrict__ M1,
                       const float* __restrict__ b1, float* __restrict__ hidden) {
  int i = blockIdx.x * blockDim.x + threadIdx.x;
  if (i >= NG * HIDC) return;
  int g = i >> 8, j = i & (HIDC - 1);
  const float* p = pooled + (size_t)g * HIDC;
  float acc = b1[j];
  for (int k = 0; k < HIDC; ++k) acc += p[k] * M1[k * HIDC + j];
  hidden[i] = acc > 0.f ? acc : 0.f;
}

__global__ void k_mlp2(const float* __restrict__ hidden, const float* __restrict__ M2,
                       const float* __restrict__ b2, float* __restrict__ out) {
  int i = blockIdx.x * blockDim.x + threadIdx.x;
  if (i >= NG * OUTC) return;
  int g = i / OUTC, o = i % OUTC;
  const float* p = hidden + (size_t)g * HIDC;
  float acc = b2[o];
  for (int k = 0; k < HIDC; ++k) acc += p[k] * M2[k * OUTC + o];
  out[i] = acc;
}

// ---------------- host ----------------
extern "C" void kernel_launch(void* const* d_in, const int* in_sizes, int n_in,
                              void* d_out, int out_size, void* d_ws, size_t ws_size,
                              hipStream_t stream) {
  const float* x    = (const float*)d_in[0];
  const int*   rows = (const int*)d_in[1];
  const int*   cols = rows + NNZ_;
  const int*   batch= (const int*)d_in[2];
  const float* W[3] = {(const float*)d_in[3], (const float*)d_in[5], (const float*)d_in[7]};
  const float* B[3] = {(const float*)d_in[4], (const float*)d_in[6], (const float*)d_in[8]};
  const float* M1   = (const float*)d_in[9];
  const float* bM1  = (const float*)d_in[10];
  const float* M2   = (const float*)d_in[11];
  const float* bM2  = (const float*)d_in[12];
  float* out = (float*)d_out;

  char* ws = (char*)d_ws;
  size_t off = 0;
  auto take = [&](size_t bytes) -> void* {
    void* p = ws + off;
    off = (off + bytes + 255) & ~(size_t)255;
    return p;
  };
  const size_t featB = (size_t)NN * HIDC * sizeof(float);
  float*    buf0   = (float*)take(featB);
  float*    buf1   = (float*)take(featB);
  float*    ebuf   = (float*)take(featB);
  float*    dinv   = (float*)take(NN * sizeof(float));
  float*    binv   = (float*)take(NE * sizeof(float));
  _Float16* Wp     = (_Float16*)take((size_t)HIDC * HIDC * sizeof(_Float16));
  float*    pooled = (float*)take((size_t)NG * HIDC * sizeof(float));
  float*    cnt    = (float*)take(64 * sizeof(float));
  float*    hidden = (float*)take((size_t)NG * HIDC * sizeof(float));

  const int ZB = 256;
  auto zero = [&](float* p, int nfloats) {
    int n4 = nfloats >> 2;
    k_zero<<<(n4 + ZB - 1) / ZB, ZB, 0, stream>>>(p, n4);
  };

  // degree inverses
  zero(dinv, NN); zero(binv, NE);
  k_count<<<(NNZ_ + 255) / 256, 256, 0, stream>>>(rows, dinv, NNZ_);
  k_count<<<(NNZ_ + 255) / 256, 256, 0, stream>>>(cols, binv, NNZ_);
  k_inv<<<(NN + 255) / 256, 256, 0, stream>>>(dinv, NN);
  k_inv<<<(NE + 255) / 256, 256, 0, stream>>>(binv, NE);

  const int elemTotal = NN * HIDC;
  auto layer = [&](const float* in, int K, const float* Wl, const float* bl,
                   float* tmp /* t and final out */) {
    // pack weights to f16 B-fragment layout
    int packN = (K >> 5) * (HIDC >> 4) * 512;
    k_pack_w<<<(packN + 255) / 256, 256, 0, stream>>>(Wl, Wp, K, HIDC);
    // t = in @ W   (WMMA, B staged through LDS)
    dim3 gg(((NN / 16) + 3) / 4, HIDC / 128);
    k_gemm<<<gg, 128, 0, stream>>>(in, Wp, tmp, NN, K, HIDC);
    // e = segsum(t[rows] -> cols) * binv
    zero(ebuf, elemTotal);
    k_scatter<<<(NNZ_ + 7) / 8, 256, 0, stream>>>(tmp, ebuf, rows, cols, NNZ_);
    k_scale_rows<<<(elemTotal + 255) / 256, 256, 0, stream>>>(ebuf, binv, NE);
    // out = segsum(e[cols] -> rows); then relu(out*dinv + b)
    zero(tmp, elemTotal);
    k_scatter<<<(NNZ_ + 7) / 8, 256, 0, stream>>>(ebuf, tmp, cols, rows, NNZ_);
    k_finish<<<(elemTotal + 255) / 256, 256, 0, stream>>>(tmp, dinv, bl, NN);
  };

  layer(x,    INC,  W[0], B[0], buf0);   // h0 in buf0
  layer(buf0, HIDC, W[1], B[1], buf1);   // h1 in buf1
  layer(buf1, HIDC, W[2], B[2], buf0);   // h2 in buf0

  // global mean pool
  zero(pooled, NG * HIDC); zero(cnt, 64);
  k_count<<<(NN + 255) / 256, 256, 0, stream>>>(batch, cnt, NN);
  k_pool<<<(NN + 7) / 8, 256, 0, stream>>>(buf0, batch, pooled);
  k_pool_div<<<(NG * HIDC + 255) / 256, 256, 0, stream>>>(pooled, cnt);

  // MLP head
  k_mlp1<<<(NG * HIDC + 255) / 256, 256, 0, stream>>>(pooled, M1, bM1, hidden);
  k_mlp2<<<(NG * OUTC + 255) / 256, 256, 0, stream>>>(hidden, M2, bM2, out);
}
